// StructuralPlasticity_79542794321996
// MI455X (gfx1250) — compile-verified, matrix operand order checked by reading, not run on hardware
//
#include <hip/hip_runtime.h>
#include <hip/hip_bf16.h>

// ---------------- problem constants ----------------
#define SLOTS 65536   // S
#define NB    1024    // B blocks
#define NN    1024    // 32*32 elements per slot

typedef __attribute__((ext_vector_type(16))) _Float16 v16h;
typedef __attribute__((ext_vector_type(8)))  float    v8f;

// -------- d_out float offsets (tuple concatenated in return order) --------
#define OFF_VIA   0           // viability          [65536]
#define OFF_THR   65536       // threshold          [1]
#define OFF_PRUNE 65537       // prune (0/1)        [65536]
#define OFF_NACT  131073      // new_active (0/1)   [65536]
#define OFF_NIND  196609      // new_in_degree      [1024]
#define OFF_NOUTD 197633      // new_out_degree     [1024]
#define OFF_NW    198657      // new_weight_values  [67108864]  (16B-misaligned!)

// -------- workspace layout (4-byte units) --------
#define WS_TSUM   0      // f32 [1024] per-row trophic sum of positives
#define WS_TMAX   1024   // f32 [1024] per-row trophic max of positives
#define WS_TCNT   2048   // i32 [1024] per-row positive count
#define WS_SACT   3072   // i32 [1]    active-slot count (atomic)
#define WS_INDEC  4096   // i32 [1024] in-degree decrements (atomic)
#define WS_OUTDEC 5120   // i32 [1024] out-degree decrements (atomic)

// =================== K0: zero the atomic counters ===================
__global__ void k0_init(int* wsI) {
    int i = blockIdx.x * blockDim.x + threadIdx.x;
    if (i == 0) wsI[WS_SACT] = 0;
    if (i < NB) { wsI[WS_INDEC + i] = 0; wsI[WS_OUTDEC + i] = 0; }
}

// =================== K1: per-slot L2 norm (WMMA reduce) + viability + copy ===
// Wave-level sum of 32 per-lane partials via V_WMMA_F32_16X16X32_F16.
// Each lane's f32 partial p is split into f16 hi/lo (p ~= hi+lo to ~2^-22 rel),
// placed at A[m][0],A[m][1] (lanes 0-15) / A[m][8],A[m][9] (lanes 16-31) per
// the ISA 16-bit A 16x32 layout. B = ones. D[m][n] = p[m] + p[m+16] (f32 acc).
__device__ __forceinline__ float wave_sum_wmma(float p) {
    _Float16 hi = (_Float16)p;
    _Float16 lo = (_Float16)(p - (float)hi);
    v16h a = {};
    a[0] = hi; a[1] = lo;
    v16h b;
#pragma unroll
    for (int i = 0; i < 16; ++i) b[i] = (_Float16)1.0f;
    v8f c = {};
    c = __builtin_amdgcn_wmma_f32_16x16x32_f16(false, a, false, b,
                                               (short)0, c, false, false);
    float t = c[0] + c[1] + c[2] + c[3] + c[4] + c[5] + c[6] + c[7];
    t += __shfl_xor(t, 16, 32);   // combine the two 16-lane halves
    return t;                     // full 32-lane sum, uniform across wave
}

__global__ void __launch_bounds__(256)
k1_viability(const float* __restrict__ wv, const float* __restrict__ tmap,
             const int* __restrict__ rows, const int* __restrict__ cols,
             const int* __restrict__ act, float* out, int* wsI) {
    const int s   = blockIdx.x;     // slot
    const int tid = threadIdx.x;    // 0..255, 8 waves, EXEC all-ones
    const float4 v = ((const float4*)(wv + (size_t)s * NN))[tid];
    // unconditional copy into new_weight_values (base is 16B-misaligned ->
    // scalar stores; loads stay b128)
    float* nw = out + OFF_NW + (size_t)s * NN + tid * 4;
    nw[0] = v.x; nw[1] = v.y; nw[2] = v.z; nw[3] = v.w;

    float p = v.x * v.x + v.y * v.y + v.z * v.z + v.w * v.w;
    float wsum = wave_sum_wmma(p);

    __shared__ float lsum[8];
    const int wid = tid >> 5, lane = tid & 31;
    if (lane == 0) lsum[wid] = wsum;
    __syncthreads();
    if (tid == 0) {
        float m2 = 0.f;
#pragma unroll
        for (int i = 0; i < 8; ++i) m2 += lsum[i];
        const float mag = sqrtf(m2);
        const int r = rows[s], cdx = cols[s];
        const float tr = tmap[(size_t)r * NB + cdx];
        const int a_ = act[s];
        out[OFF_VIA + s] = a_ ? mag * (1.0f + tr) : 0.0f;
        if (a_) atomicAdd(&wsI[WS_SACT], 1);
    }
}

// =================== K2: trophic-map per-row partials ===================
__global__ void __launch_bounds__(256)
k2_trophic(const float* __restrict__ tmap, float* wsF, int* wsI) {
    const int row = blockIdx.x, tid = threadIdx.x;
    const float4 v = ((const float4*)(tmap + (size_t)row * NB))[tid];
    float s = 0.f, mx = 0.f; int c = 0;
    if (v.x > 0.f) { s += v.x; mx = fmaxf(mx, v.x); ++c; }
    if (v.y > 0.f) { s += v.y; mx = fmaxf(mx, v.y); ++c; }
    if (v.z > 0.f) { s += v.z; mx = fmaxf(mx, v.z); ++c; }
    if (v.w > 0.f) { s += v.w; mx = fmaxf(mx, v.w); ++c; }
    __shared__ float ss[256], sm[256];
    __shared__ int   sc[256];
    ss[tid] = s; sm[tid] = mx; sc[tid] = c;
    __syncthreads();
    for (int st = 128; st > 0; st >>= 1) {
        if (tid < st) {
            ss[tid] += ss[tid + st];
            sm[tid] = fmaxf(sm[tid], sm[tid + st]);
            sc[tid] += sc[tid + st];
        }
        __syncthreads();
    }
    if (tid == 0) {
        wsF[WS_TSUM + row] = ss[0];
        wsF[WS_TMAX + row] = sm[0];
        wsI[WS_TCNT + row] = sc[0];
    }
}

// =================== K3: quantile threshold (radix-histogram select) =========
// Exact k-th order statistic of positive f32 viabilities: positive floats are
// order-isomorphic to their uint bit patterns; 3 LDS-histogram rounds over
// bits 30:20, 19:9, 8:0 pin the key exactly. Deterministic (int atomics only).
__device__ unsigned select_kth(const float* __restrict__ via, int rank,
                               int tid, int* hist, int* shI) {
    unsigned prefix = 0u;
    unsigned hmask  = 0x80000000u;   // sign bit (always 0 for v > 0)
    if (tid == 0) shI[0] = rank;
    __syncthreads();
    const int shifts[3] = {20, 9, 0};
    const int bitsv[3]  = {11, 11, 9};
    for (int rd = 0; rd < 3; ++rd) {
        const int shift = shifts[rd];
        const int bins  = 1 << bitsv[rd];
        const unsigned bmask = (unsigned)(bins - 1);
        for (int i = tid; i < bins; i += 1024) hist[i] = 0;
        __syncthreads();
        for (int j = tid; j < SLOTS; j += 1024) {
            const float v = via[j];
            if (v > 0.0f) {
                const unsigned k = __float_as_uint(v);
                if ((k & hmask) == prefix)
                    atomicAdd(&hist[(int)((k >> shift) & bmask)], 1);
            }
        }
        __syncthreads();
        if (tid == 0) {
            int rr = shI[0];
            int b = 0;
            for (; b < bins - 1; ++b) {
                const int c = hist[b];
                if (rr < c) break;
                rr -= c;
            }
            shI[0] = rr;
            shI[1] = b;
        }
        __syncthreads();
        prefix |= ((unsigned)shI[1]) << shift;
        hmask  |= (bmask << shift);
        __syncthreads();
    }
    return prefix;
}

__global__ void __launch_bounds__(1024)
k3_threshold(float* out, const float* wsF, const int* wsI) {
    const int tid = threadIdx.x;
    __shared__ float fred[1024];
    __shared__ int   ired[1024];
    __shared__ int   hist[2048];
    __shared__ int   shI[4];   // [0]=rank scratch [1]=bin [2]=m [3]=tcnt
    __shared__ float shF[2];   // [0]=tsum [1]=tmax
    const float* via = out + OFF_VIA;

    fred[tid] = wsF[WS_TSUM + tid];
    __syncthreads();
    for (int st = 512; st > 0; st >>= 1) { if (tid < st) fred[tid] += fred[tid + st]; __syncthreads(); }
    if (tid == 0) shF[0] = fred[0];
    __syncthreads();
    fred[tid] = wsF[WS_TMAX + tid];
    __syncthreads();
    for (int st = 512; st > 0; st >>= 1) { if (tid < st) fred[tid] = fmaxf(fred[tid], fred[tid + st]); __syncthreads(); }
    if (tid == 0) shF[1] = fred[0];
    __syncthreads();
    ired[tid] = wsI[WS_TCNT + tid];
    __syncthreads();
    for (int st = 512; st > 0; st >>= 1) { if (tid < st) ired[tid] += ired[tid + st]; __syncthreads(); }
    if (tid == 0) shI[3] = ired[0];
    __syncthreads();
    int cnt = 0;
    for (int j = tid; j < SLOTS; j += 1024) cnt += (via[j] > 0.0f) ? 1 : 0;
    ired[tid] = cnt;
    __syncthreads();
    for (int st = 512; st > 0; st >>= 1) { if (tid < st) ired[tid] += ired[tid + st]; __syncthreads(); }
    if (tid == 0) shI[2] = ired[0];
    __syncthreads();

    const int   m    = shI[2];
    const float tsum = shF[0];
    const float tmax = shF[1];
    const int   tcnt = shI[3];
    const int   sact = wsI[WS_SACT];

    // survival percentile (all uniform across the block)
    const float scarcity = (float)sact / (float)SLOTS;   // 1 - num_free/S
    float demand = 0.0f;
    if (tcnt > 0) {
        const float avg = tsum / fmaxf((float)tcnt, 1.0f);
        demand = fminf(fmaxf(avg / (tmax + 1e-8f), 0.0f), 1.0f);
    }
    const float  sp = 10.0f + scarcity * 25.0f + demand * 15.0f;
    const double q  = (double)sp / 100.0;

    float threshold = 0.0f;
    if (m > 0) {                               // uniform branch (barriers inside ok)
        double pos = q * (double)(m - 1);
        int i0 = (int)pos;
        if (i0 > m - 1) i0 = m - 1;
        const double frac = pos - (double)i0;
        const float s0 = __uint_as_float(select_kth(via, i0, tid, hist, shI));
        float s1 = s0;
        if (i0 + 1 <= m - 1)
            s1 = __uint_as_float(select_kth(via, i0 + 1, tid, hist, shI));
        threshold = s0 + (float)frac * (s1 - s0);
    }
    if (tid == 0) out[OFF_THR] = threshold;
}

// =================== K4: prune decision + degree decrements ===================
__global__ void __launch_bounds__(256)
k4_prune(float* out, const int* __restrict__ rows, const int* __restrict__ cols,
         const int* __restrict__ act, const int* __restrict__ indeg,
         const int* __restrict__ outdeg, int* wsI) {
    const int s = blockIdx.x * blockDim.x + threadIdx.x;
    if (s >= SLOTS) return;
    const float via = out[OFF_VIA + s];
    const float thr = out[OFF_THR];
    const int a_ = act[s];
    const int r = rows[s], cdx = cols[s];
    const bool prune = (a_ != 0) && (via < thr) && (r != cdx) &&
                       (indeg[cdx] > 1) && (outdeg[r] > 1);
    out[OFF_PRUNE + s] = prune ? 1.0f : 0.0f;
    out[OFF_NACT  + s] = ((a_ != 0) && !prune) ? 1.0f : 0.0f;
    if (prune) {
        atomicAdd(&wsI[WS_INDEC  + cdx], 1);
        atomicAdd(&wsI[WS_OUTDEC + r  ], 1);
    }
}

// =================== K5: finalize degrees ===================
__global__ void k5_degrees(const int* __restrict__ indeg,
                           const int* __restrict__ outdeg,
                           const int* wsI, float* out) {
    const int b = blockIdx.x * blockDim.x + threadIdx.x;
    if (b < NB) {
        out[OFF_NIND  + b] = (float)(indeg[b]  - wsI[WS_INDEC  + b]);
        out[OFF_NOUTD + b] = (float)(outdeg[b] - wsI[WS_OUTDEC + b]);
    }
}

// =================== K6: zero pruned slots' weight blocks ===================
__global__ void __launch_bounds__(256)
k6_zero(float* out) {
    const int s = blockIdx.x;
    if (out[OFF_PRUNE + s] == 0.0f) return;   // uniform early-exit
    const int tid = threadIdx.x;
    float* nw = out + OFF_NW + (size_t)s * NN + tid * 4;
    nw[0] = 0.f; nw[1] = 0.f; nw[2] = 0.f; nw[3] = 0.f;
}

// =================== launcher ===================
extern "C" void kernel_launch(void* const* d_in, const int* in_sizes, int n_in,
                              void* d_out, int out_size, void* d_ws, size_t ws_size,
                              hipStream_t stream) {
    const float* wv     = (const float*)d_in[0];   // weight_values  [S,32,32]
    const float* tmap   = (const float*)d_in[1];   // trophic map    [B,B]
    const int*   rows   = (const int*)  d_in[2];   // weight_rows    [S]
    const int*   cols   = (const int*)  d_in[3];   // weight_cols    [S]
    const int*   act    = (const int*)  d_in[4];   // active_blocks  [S]
    const int*   indeg  = (const int*)  d_in[5];   // in_degree      [B]
    const int*   outdeg = (const int*)  d_in[6];   // out_degree     [B]
    float* out = (float*)d_out;
    int*   wsI = (int*)  d_ws;
    float* wsF = (float*)d_ws;

    k0_init     <<<4, 256, 0, stream>>>(wsI);
    k1_viability<<<SLOTS, 256, 0, stream>>>(wv, tmap, rows, cols, act, out, wsI);
    k2_trophic  <<<NB, 256, 0, stream>>>(tmap, wsF, wsI);
    k3_threshold<<<1, 1024, 0, stream>>>(out, wsF, wsI);
    k4_prune    <<<SLOTS / 256, 256, 0, stream>>>(out, rows, cols, act, indeg, outdeg, wsI);
    k5_degrees  <<<4, 256, 0, stream>>>(indeg, outdeg, wsI, out);
    k6_zero     <<<SLOTS, 256, 0, stream>>>(out);
}
